// PKM_2027224564000
// MI455X (gfx1250) — matrix operand order repaired and checked by reading, not compiled, not run on hardware
//
#include <hip/hip_runtime.h>
#include <hip/hip_bf16.h>
#include <stdint.h>

// ---------- types ----------
typedef __attribute__((ext_vector_type(8)))  __bf16 v8bf;
typedef __attribute__((ext_vector_type(16))) __bf16 v16bf;
typedef __attribute__((ext_vector_type(8)))  float  v8f;

// ---------- bf16 helpers ----------
__device__ __forceinline__ unsigned short bf16_rn(float f) {
  unsigned u = __builtin_bit_cast(unsigned, f);
  u += 0x7FFFu + ((u >> 16) & 1u);            // round-to-nearest-even
  return (unsigned short)(u >> 16);
}
__device__ __forceinline__ float bf16_to_f(unsigned short b) {
  return __builtin_bit_cast(float, ((unsigned)b) << 16);
}
__device__ __forceinline__ unsigned pack2(unsigned short a, unsigned short b) {
  return (unsigned)a | ((unsigned)b << 16);
}
// load 16 bf16 from two contiguous 8-element (16B) chunks
__device__ __forceinline__ v16bf ld16(const unsigned short* p0, const unsigned short* p1) {
  v8bf a = *(const v8bf*)p0;
  v8bf b = *(const v8bf*)p1;
  return __builtin_shufflevector(a, b, 0,1,2,3,4,5,6,7,8,9,10,11,12,13,14,15);
}
__device__ __forceinline__ v8f wmma3(v8f c, v16bf ah, v16bf al, v16bf bh, v16bf bl) {
  // C += Ah*Bh + Ah*Bl + Al*Bh  (near-fp32 precision with bf16 WMMA)
  c = __builtin_amdgcn_wmma_f32_16x16x32_bf16(false, ah, false, bh, (short)0, c, false, false);
  c = __builtin_amdgcn_wmma_f32_16x16x32_bf16(false, ah, false, bl, (short)0, c, false, false);
  c = __builtin_amdgcn_wmma_f32_16x16x32_bf16(false, al, false, bh, (short)0, c, false, false);
  return c;
}

// ---------- Kernel 0: f32 -> packed hi/lo bf16 (row-major preserved) ----------
__global__ __launch_bounds__(256) void pkm_split(const float* __restrict__ src,
                                                 unsigned* __restrict__ hi,   // 2 bf16 per word
                                                 unsigned* __restrict__ lo,
                                                 int n4) {
  const int i = blockIdx.x * 256 + threadIdx.x;
  if (i >= n4) return;
  const float4 f = ((const float4*)src)[i];
  const unsigned short h0 = bf16_rn(f.x), h1 = bf16_rn(f.y), h2 = bf16_rn(f.z), h3 = bf16_rn(f.w);
  const unsigned short l0 = bf16_rn(f.x - bf16_to_f(h0));
  const unsigned short l1 = bf16_rn(f.y - bf16_to_f(h1));
  const unsigned short l2 = bf16_rn(f.z - bf16_to_f(h2));
  const unsigned short l3 = bf16_rn(f.w - bf16_to_f(h3));
  hi[i*2+0] = pack2(h0, h1); hi[i*2+1] = pack2(h2, h3);
  lo[i*2+0] = pack2(l0, l1); lo[i*2+1] = pack2(l2, l3);
}

// ---------- Kernel 0b: keys [h,n,p,d] -> hi/lo bf16 reordered to [h,p,n,d] ----------
__global__ __launch_bounds__(256) void pkm_split_keys(const float* __restrict__ K,
                                                      unsigned* __restrict__ hi,
                                                      unsigned* __restrict__ lo) {
  const int i = blockIdx.x * 256 + threadIdx.x;   // pair index, 262144 total
  const int d2 = i & 63;                          // d = d2*2
  const int rest = i >> 6;
  const int p = rest & 1;
  const int n = (rest >> 1) & 255;
  const int h = rest >> 9;
  const float2 f = ((const float2*)K)[i];
  const unsigned short h0 = bf16_rn(f.x), h1 = bf16_rn(f.y);
  const unsigned short l0 = bf16_rn(f.x - bf16_to_f(h0));
  const unsigned short l1 = bf16_rn(f.y - bf16_to_f(h1));
  const int out = (((h*2 + p)*256 + n)*64) + d2;  // in packed-pair units
  hi[out] = pack2(h0, h1);
  lo[out] = pack2(l0, l1);
}

// ---------- Kernel 1: q = x @ Wq^T  (M=2048, N=2048, K=512), bf16x3 operands ----------
__global__ __launch_bounds__(256) void pkm_gemm_q(const unsigned short* __restrict__ Xh,
                                                  const unsigned short* __restrict__ Xl,
                                                  const unsigned short* __restrict__ Wh,
                                                  const unsigned short* __restrict__ Wl,
                                                  float* __restrict__ Q) {
  const int lane  = threadIdx.x & 31;
  const int wave  = threadIdx.x >> 5;
  const int tile  = blockIdx.x * 8 + wave;   // 16384 tiles
  const int tn    = (tile & 127) << 4;
  const int tm    = (tile >> 7) << 4;
  const int l15   = lane & 15;
  const int lhalf = lane >> 4;
  const size_t aro = (size_t)(tm + l15) * 512;
  const size_t bro = (size_t)(tn + l15) * 512;
  v8f c = {0.f,0.f,0.f,0.f,0.f,0.f,0.f,0.f};
  for (int kk = 0; kk < 512; kk += 32) {
    const int ao = kk + lhalf*8;    // A: e<8 -> K=ao+e ; e>=8 -> K=ao+16+(e-8)
    const int bo = kk + lhalf*8;    // B: e<8 -> K=bo+e ; e>=8 -> K=bo+16+(e-8)
    v16bf ah = ld16(Xh + aro + ao, Xh + aro + ao + 16);
    v16bf al = ld16(Xl + aro + ao, Xl + aro + ao + 16);
    v16bf bh = ld16(Wh + bro + bo, Wh + bro + bo + 16);
    v16bf bl = ld16(Wl + bro + bo, Wl + bro + bo + 16);
    c = wmma3(c, ah, al, bh, bl);
  }
  #pragma unroll
  for (int r = 0; r < 8; ++r)
    Q[(size_t)(tm + r + lhalf*8) * 2048 + tn + l15] = c[r];
}

// ---------- Kernel 2: LayerNorm(128) -> hi/lo bf16 q_norm ----------
__global__ __launch_bounds__(256) void pkm_ln(const float* __restrict__ Q,
                                              const float* __restrict__ gamma,
                                              const float* __restrict__ beta,
                                              unsigned short* __restrict__ QNh,
                                              unsigned short* __restrict__ QNl) {
  const int lane = threadIdx.x & 31;
  const int wave = threadIdx.x >> 5;
  const int row  = blockIdx.x * 8 + wave;    // 32768 rows of 128
  const float4 v0 = *(const float4*)(Q + (size_t)row * 128 + lane * 4);
  float s  = v0.x + v0.y + v0.z + v0.w;
  float s2 = v0.x*v0.x + v0.y*v0.y + v0.z*v0.z + v0.w*v0.w;
  #pragma unroll
  for (int off = 16; off >= 1; off >>= 1) {
    s  += __shfl_xor(s,  off, 32);
    s2 += __shfl_xor(s2, off, 32);
  }
  const float mu  = s * (1.f / 128.f);
  const float var = s2 * (1.f / 128.f) - mu * mu;
  const float inv = rsqrtf(var + 1e-5f);
  const float4 g = *(const float4*)(gamma + lane * 4);
  const float4 b = *(const float4*)(beta  + lane * 4);
  float r0 = (v0.x - mu) * inv * g.x + b.x;
  float r1 = (v0.y - mu) * inv * g.y + b.y;
  float r2 = (v0.z - mu) * inv * g.z + b.z;
  float r3 = (v0.w - mu) * inv * g.w + b.w;
  const unsigned short h0 = bf16_rn(r0), h1 = bf16_rn(r1), h2 = bf16_rn(r2), h3 = bf16_rn(r3);
  uint2 H, L;
  H.x = pack2(h0, h1); H.y = pack2(h2, h3);
  L.x = pack2(bf16_rn(r0 - bf16_to_f(h0)), bf16_rn(r1 - bf16_to_f(h1)));
  L.y = pack2(bf16_rn(r2 - bf16_to_f(h2)), bf16_rn(r3 - bf16_to_f(h3)));
  *(uint2*)(QNh + (size_t)row * 128 + lane * 4) = H;
  *(uint2*)(QNl + (size_t)row * 128 + lane * 4) = L;
}

// ---------- Kernel 3: dots[tok,h,p,n] = q_norm . keys ----------
__global__ __launch_bounds__(256) void pkm_gemm_dots(const unsigned short* __restrict__ QNh,
                                                     const unsigned short* __restrict__ QNl,
                                                     const unsigned short* __restrict__ Kh,
                                                     const unsigned short* __restrict__ Kl,
                                                     float* __restrict__ Dots) {
  const int lane  = threadIdx.x & 31;
  const int wave  = threadIdx.x >> 5;
  const int hp    = blockIdx.y;              // h*2+p, 16 combos
  const int h     = hp >> 1, p = hp & 1;
  const int tile  = blockIdx.x * 8 + wave;   // 2048 tiles (128 M x 16 N)
  const int tn    = (tile & 15) << 4;
  const int tm    = (tile >> 4) << 4;
  const int l15   = lane & 15;
  const int lhalf = lane >> 4;
  const size_t aro = (size_t)p*1024 + h*128 + (size_t)(tm + l15) * 2048;
  const size_t bro = ((size_t)hp * 256 + tn + l15) * 128;
  v8f c = {0.f,0.f,0.f,0.f,0.f,0.f,0.f,0.f};
  #pragma unroll
  for (int kk = 0; kk < 128; kk += 32) {
    const int ao = kk + lhalf*8;
    const int bo = kk + lhalf*8;
    v16bf ah = ld16(QNh + aro + ao, QNh + aro + ao + 16);
    v16bf al = ld16(QNl + aro + ao, QNl + aro + ao + 16);
    v16bf bh = ld16(Kh + bro + bo, Kh + bro + bo + 16);
    v16bf bl = ld16(Kl + bro + bo, Kl + bro + bo + 16);
    c = wmma3(c, ah, al, bh, bl);
  }
  #pragma unroll
  for (int r = 0; r < 8; ++r)
    Dots[((size_t)(tm + r + lhalf*8) * 16 + hp) * 256 + tn + l15] = c[r];
}

// ---------- top-32 of 256 with one wave ----------
__device__ __forceinline__ void select_top32(const float* __restrict__ src, int lane,
                                             float& s_out, int& i_out) {
  float v[8];
  #pragma unroll
  for (int i = 0; i < 8; ++i) v[i] = src[lane + 32 * i];
  s_out = 0.f; i_out = 0;
  for (int k = 0; k < 32; ++k) {
    float m = -3.4e38f; int mi = 0x7fffffff;
    #pragma unroll
    for (int i = 0; i < 8; ++i) {
      int n = lane + 32 * i;
      if (v[i] > m || (v[i] == m && n < mi)) { m = v[i]; mi = n; }
    }
    #pragma unroll
    for (int off = 16; off >= 1; off >>= 1) {
      float om = __shfl_xor(m, off, 32);
      int omi  = __shfl_xor(mi, off, 32);
      if (om > m || (om == m && omi < mi)) { m = om; mi = omi; }
    }
    if (lane == (mi & 31)) v[mi >> 5] = -3.4e38f;
    if (lane == k) { s_out = m; i_out = mi; }
  }
}

// ---------- Kernel 4: two-stage top-k + softmax; one wave per (token, head) ----------
__global__ __launch_bounds__(256) void pkm_topk(const float* __restrict__ Dots,
                                                float* __restrict__ Wout,
                                                int* __restrict__ VIout) {
  const int lane = threadIdx.x & 31;
  const int wave = threadIdx.x >> 5;
  const int g = blockIdx.x * 8 + wave;       // token*8 + head, 16384 groups
  const float* base = Dots + (size_t)g * 2 * 256;
  float sx, sy; int ix, iy;
  select_top32(base,        lane, sx, ix);   // p = 0
  select_top32(base + 256,  lane, sy, iy);   // p = 1
  float cand[32];
  #pragma unroll
  for (int j = 0; j < 32; ++j) cand[j] = sx + __shfl(sy, j, 32);
  unsigned used = 0;
  float fs = 0.f; int fi = 0;
  for (int k = 0; k < 32; ++k) {
    float m = -3.4e38f; int mid = 0x7fffffff;
    #pragma unroll
    for (int j = 0; j < 32; ++j) {
      bool avail = ((used >> j) & 1u) == 0u;
      float cv = avail ? cand[j] : -3.4e38f;
      int cid = (lane << 5) | j;
      if (cv > m || (cv == m && cid < mid)) { m = cv; mid = cid; }
    }
    #pragma unroll
    for (int off = 16; off >= 1; off >>= 1) {
      float om = __shfl_xor(m, off, 32);
      int omid = __shfl_xor(mid, off, 32);
      if (om > m || (om == m && omid < mid)) { m = om; mid = omid; }
    }
    if ((mid >> 5) == lane) used |= 1u << (mid & 31);
    if (lane == k) { fs = m; fi = mid; }
  }
  const int vi = __shfl(ix, fi >> 5, 32) * 256 + __shfl(iy, fi & 31, 32);
  const float mx = __shfl(fs, 0, 32);
  const float e = __expf(fs - mx);
  float ssum = e;
  #pragma unroll
  for (int off = 16; off >= 1; off >>= 1) ssum += __shfl_xor(ssum, off, 32);
  Wout[(size_t)g * 32 + lane]  = e / ssum;
  VIout[(size_t)g * 32 + lane] = vi;
}

// ---------- Kernel 5: out[tok,:] = sum_e w[e] * values[vi[e],:]  (values L2-resident) ----------
__global__ __launch_bounds__(256) void pkm_gather(const float* __restrict__ Values,
                                                  const float* __restrict__ W,
                                                  const int* __restrict__ VI,
                                                  float* __restrict__ Out) {
  __shared__ float ws_[256];
  __shared__ int   vs_[256];
  const int tok = blockIdx.x;
  const int tid = threadIdx.x;
  ws_[tid] = W[(size_t)tok * 256 + tid];
  vs_[tid] = VI[(size_t)tok * 256 + tid];
  __syncthreads();
  float acc0 = 0.f, acc1 = 0.f;
  for (int e = 0; e < 256; ++e) {
    const float* vr = Values + (size_t)vs_[e] * 512;
    const float w = ws_[e];
    acc0 += w * vr[tid];
    acc1 += w * vr[tid + 256];
  }
  Out[(size_t)tok * 512 + tid]       = acc0;
  Out[(size_t)tok * 512 + tid + 256] = acc1;
}

// ---------- launch ----------
extern "C" void kernel_launch(void* const* d_in, const int* in_sizes, int n_in,
                              void* d_out, int out_size, void* d_ws, size_t ws_size,
                              hipStream_t stream) {
  const float* x      = (const float*)d_in[0];  // [2,1024,512]
  const float* Wq     = (const float*)d_in[1];  // [2048,512]
  const float* keys   = (const float*)d_in[2];  // [8,256,2,128]
  const float* values = (const float*)d_in[3];  // [65536,512]
  const float* gamma  = (const float*)d_in[4];  // [128]
  const float* beta   = (const float*)d_in[5];  // [128]

  char* ws = (char*)d_ws;
  float*          q    = (float*)          (ws);               // 16 MB
  float*          dots = (float*)          (ws + (16u << 20)); // 16 MB
  float*          wts  = (float*)          (ws + (32u << 20)); //  2 MB
  int*            vidx = (int*)            (ws + (34u << 20)); //  2 MB
  unsigned short* xh   = (unsigned short*) (ws + (36u << 20)); //  2 MB
  unsigned short* xl   = (unsigned short*) (ws + (38u << 20)); //  2 MB
  unsigned short* wqh  = (unsigned short*) (ws + (40u << 20)); //  2 MB
  unsigned short* wql  = (unsigned short*) (ws + (42u << 20)); //  2 MB
  unsigned short* kh   = (unsigned short*) (ws + (44u << 20)); //  1 MB
  unsigned short* kl   = (unsigned short*) (ws + (45u << 20)); //  1 MB
  unsigned short* qnh  = (unsigned short*) (ws + (46u << 20)); //  8 MB
  unsigned short* qnl  = (unsigned short*) (ws + (54u << 20)); //  8 MB

  pkm_split     <<<1024, 256, 0, stream>>>(x,  (unsigned*)xh,  (unsigned*)xl,  262144);
  pkm_split     <<<1024, 256, 0, stream>>>(Wq, (unsigned*)wqh, (unsigned*)wql, 262144);
  pkm_split_keys<<<1024, 256, 0, stream>>>(keys, (unsigned*)kh, (unsigned*)kl);
  pkm_gemm_q    <<<2048, 256, 0, stream>>>(xh, xl, wqh, wql, q);
  pkm_ln        <<<4096, 256, 0, stream>>>(q, gamma, beta, qnh, qnl);
  pkm_gemm_dots <<<dim3(256, 16), 256, 0, stream>>>(qnh, qnl, kh, kl, dots);
  pkm_topk      <<<2048, 256, 0, stream>>>(dots, wts, vidx);
  pkm_gather    <<<2048, 256, 0, stream>>>(values, wts, vidx, (float*)d_out);
}
